// CLIPAttentionSVD_36644660970258
// MI455X (gfx1250) — matrix operand order repaired and checked by reading, not compiled, not run on hardware
//
#include <hip/hip_runtime.h>

// ---------------------------------------------------------------------------
// CDNA5 / gfx1250 implementation of dual low-rank+full-rank CLIP attention.
// All matmuls run on v_wmma_f32_16x16x32_bf16 (wave32, 16x16x32 tiles).
// GEMM tile movement uses GLOBAL_LOAD_ASYNC_TO_LDS_B128 (ASYNCcnt) with a
// two-stage LDS double buffer.
// ---------------------------------------------------------------------------

typedef __attribute__((ext_vector_type(16))) __bf16 v16bf;
typedef __attribute__((ext_vector_type(8)))  float  v8f;

namespace {
constexpr int kB  = 64;
constexpr int kN  = 257;
constexpr int kC  = 1024;
constexpr int kH  = 16;
constexpr int kM  = kB * kN;      // 16448 tokens
constexpr int kNQ = 272;          // queries padded to 17*16
constexpr int kNK = 288;          // keys padded to 9*32
constexpr int kF  = 96;           // 16 (lr) + 64 (full) padded to 3*32
constexpr int kQKVN = 3840;       // 3*256 + 3*1024 fused projection outputs
constexpr int kK2   = 1280;       // H * 80 concat for output projection
constexpr float kScale = 0.125f;  // (C/H)^-0.5 = 64^-0.5
}

// ---------------- bf16 helpers ---------------------------------------------
__device__ __forceinline__ unsigned short f2bf(float x) {
  unsigned int u = __float_as_uint(x);
  u += 0x7fffu + ((u >> 16) & 1u);           // round-to-nearest-even
  return (unsigned short)(u >> 16);
}

union FragU { v16bf v; uint4 u[2]; };

// A fragment (16x32, bf16): lane<16 -> M=lane, K in {0..7, 16..23};
// lane>=16 -> M=lane-16, K in {8..15, 24..31}. Two 16B loads per lane.
__device__ __forceinline__ v16bf load_frag_a(const unsigned short* base,
                                             int stride, int lane) {
  const int m  = lane & 15;
  const int kh = (lane >> 4) << 3;           // 0 or 8
  const unsigned short* p = base + m * stride + kh;
  FragU f;
  f.u[0] = *(const uint4*)(p);
  f.u[1] = *(const uint4*)(p + 16);
  return f.v;
}

// B fragment (32x16, bf16): lane<16 -> N=lane, K in {0..7, 16..23};
// lane>=16 -> N=lane-16, K in {8..15, 24..31}. Two 16B loads per lane.
__device__ __forceinline__ v16bf load_frag_b(const unsigned short* base,
                                             int stride, int lane) {
  const int n  = lane & 15;
  const int k0 = (lane >> 4) << 3;           // 0 or 8
  const unsigned short* p = base + n * stride + k0;
  FragU f;
  f.u[0] = *(const uint4*)(p);
  f.u[1] = *(const uint4*)(p + 16);
  return f.v;
}

__device__ __forceinline__ v8f wmma_bf16(v16bf a, v16bf b, v8f c) {
  return __builtin_amdgcn_wmma_f32_16x16x32_bf16(
      /*neg_a=*/false, a, /*neg_b=*/false, b,
      /*c_mod=*/(short)0, c, /*reuse_a=*/false, /*reuse_b=*/false);
}

// ---------------- async memory -> LDS (CDNA5, ASYNCcnt) --------------------
// Generic pointers to LDS on amdgcn carry the LDS byte offset in the low 32
// bits (ISA 10.2 aperture rules), which is exactly what the async-load VDST
// operand wants (an LDS byte address, plus the instruction offset).
__device__ __forceinline__ unsigned lds_off(const void* p) {
  return (unsigned)(unsigned long long)p;
}

__device__ __forceinline__ void async_load_b128(unsigned lds_addr,
                                                const void* gaddr) {
  asm volatile("global_load_async_to_lds_b128 %0, %1, off"
               :: "v"(lds_addr), "v"(gaddr)
               : "memory");
}

#define WAIT_ASYNC_4() asm volatile("s_wait_asynccnt 0x4" ::: "memory")
#define WAIT_ASYNC_0() asm volatile("s_wait_asynccnt 0x0" ::: "memory")

// ---------------- utility kernels ------------------------------------------
__global__ void k_zero_u4(uint4* __restrict__ p, long n) {
  long i = (long)blockIdx.x * blockDim.x + threadIdx.x;
  if (i < n) p[i] = make_uint4(0u, 0u, 0u, 0u);
}

__global__ void k_cvt_bf16(const float* __restrict__ src,
                           unsigned short* __restrict__ dst, long n) {
  long i = (long)blockIdx.x * blockDim.x + threadIdx.x;
  if (i < n) dst[i] = f2bf(src[i]);
}

// Fuse six projection weights into one [3840, 1024] bf16 matrix (row = out).
__global__ void k_pack_wqkv(const float* __restrict__ wq_lr,
                            const float* __restrict__ wk_lr,
                            const float* __restrict__ wv_lr,
                            const float* __restrict__ wq_f,
                            const float* __restrict__ wk_f,
                            const float* __restrict__ wv_f,
                            unsigned short* __restrict__ out) {
  long i = (long)blockIdx.x * blockDim.x + threadIdx.x;
  if (i >= (long)kQKVN * kC) return;
  int j = (int)(i / kC);
  int k = (int)(i % kC);
  float v;
  if      (j <  256) v = wq_lr[(size_t)j          * kC + k];
  else if (j <  512) v = wk_lr[(size_t)(j -  256) * kC + k];
  else if (j <  768) v = wv_lr[(size_t)(j -  512) * kC + k];
  else if (j < 1792) v = wq_f [(size_t)(j -  768) * kC + k];
  else if (j < 2816) v = wk_f [(size_t)(j - 1792) * kC + k];
  else               v = wv_f [(size_t)(j - 2816) * kC + k];
  out[i] = f2bf(v);
}

// Fuse Wo_lr [1024,256] and Wo_full [1024,1024] into Wo_cat [1024,1280]
// whose column h*80+c maps to (c<16 ? Wo_lr[:,h*16+c] : Wo_full[:,h*64+c-16]).
__global__ void k_pack_wo(const float* __restrict__ wo_lr,
                          const float* __restrict__ wo_f,
                          unsigned short* __restrict__ out) {
  long i = (long)blockIdx.x * blockDim.x + threadIdx.x;
  if (i >= (long)kC * kK2) return;
  int o  = (int)(i / kK2);
  int c  = (int)(i % kK2);
  int h  = c / 80;
  int cc = c % 80;
  float v = (cc < 16) ? wo_lr[(size_t)o * 256  + h * 16 + cc]
                      : wo_f [(size_t)o * kC   + h * 64 + (cc - 16)];
  out[i] = f2bf(v);
}

// ---------------- tiled WMMA GEMM mainloop ---------------------------------
// Block = 256 threads = 8 waves; computes a 128(M) x 128(N) tile.
// Wave (wm,wn) owns a 32x64 region = 2x4 WMMA 16x16 tiles.
// Double-buffered LDS fed by async loads: issue stage n+1's 4 loads/thread,
// s_wait_asynccnt 4 completes stage n (async loads finish in order).
// Out-of-range A rows are clamped (row r only feeds discarded output row r).
__device__ __forceinline__ void gemm_load_stage(
    const unsigned short* __restrict__ X, int Mclamp, int K,
    const unsigned short* __restrict__ W, int m0, int n0, int k0,
    unsigned short (*As)[40], unsigned short (*Bs)[40]) {
  const int t = threadIdx.x;
#pragma unroll
  for (int it = 0; it < 2; ++it) {
    int r  = (t >> 2) + it * 64;            // 0..127
    int c8 = (t & 3) << 3;                  // 0,8,16,24
    int gm = m0 + r;
    if (gm > Mclamp) gm = Mclamp;
    async_load_b128(lds_off(&As[r][c8]), X + (size_t)gm * K + k0 + c8);
    async_load_b128(lds_off(&Bs[r][c8]), W + (size_t)(n0 + r) * K + k0 + c8);
  }
}

__device__ __forceinline__ void gemm_mainloop(
    const unsigned short* __restrict__ X, int Mreal, int K,
    const unsigned short* __restrict__ W, int m0, int n0,
    unsigned short (*As)[128][40], unsigned short (*Bs)[128][40],
    v8f acc[2][4]) {
  const int t    = threadIdx.x;
  const int w    = t >> 5;
  const int lane = t & 31;
  const int wm   = w >> 1;          // 0..3 -> 32-row band
  const int wn   = w & 1;           // 0..1 -> 64-col band
  const int Mclamp = Mreal - 1;

  gemm_load_stage(X, Mclamp, K, W, m0, n0, 0, As[0], Bs[0]);

  for (int k0 = 0; k0 < K; k0 += 32) {
    const int cur = (k0 >> 5) & 1;
    if (k0 + 32 < K) {
      gemm_load_stage(X, Mclamp, K, W, m0, n0, k0 + 32, As[cur ^ 1],
                      Bs[cur ^ 1]);
      WAIT_ASYNC_4();                       // current stage's 4 loads done
    } else {
      WAIT_ASYNC_0();
    }
    __syncthreads();

    v16bf a0 = load_frag_a(&As[cur][wm * 32 + 0][0],  40, lane);
    v16bf a1 = load_frag_a(&As[cur][wm * 32 + 16][0], 40, lane);
    v16bf b0 = load_frag_b(&Bs[cur][wn * 64 + 0][0],  40, lane);
    v16bf b1 = load_frag_b(&Bs[cur][wn * 64 + 16][0], 40, lane);
    v16bf b2 = load_frag_b(&Bs[cur][wn * 64 + 32][0], 40, lane);
    v16bf b3 = load_frag_b(&Bs[cur][wn * 64 + 48][0], 40, lane);

    acc[0][0] = wmma_bf16(a0, b0, acc[0][0]);
    acc[0][1] = wmma_bf16(a0, b1, acc[0][1]);
    acc[0][2] = wmma_bf16(a0, b2, acc[0][2]);
    acc[0][3] = wmma_bf16(a0, b3, acc[0][3]);
    acc[1][0] = wmma_bf16(a1, b0, acc[1][0]);
    acc[1][1] = wmma_bf16(a1, b1, acc[1][1]);
    acc[1][2] = wmma_bf16(a1, b2, acc[1][2]);
    acc[1][3] = wmma_bf16(a1, b3, acc[1][3]);
    __syncthreads();
  }
}

// ---------------- stage 1: fused QKV GEMM with scatter epilogue ------------
// qcat: [B*H][272][96] bf16   (cols 0..15 = lr, 16..79 = full, 80..95 = 0)
// kcat: [B*H][288][96] bf16
// vT  : [B*H][96][288] bf16   (transposed so AV B-frags read contiguous K)
__global__ __launch_bounds__(256) void k_gemm_qkv(
    const unsigned short* __restrict__ X, const unsigned short* __restrict__ W,
    unsigned short* __restrict__ qcat, unsigned short* __restrict__ kcat,
    unsigned short* __restrict__ vT) {
  __shared__ unsigned short As[2][128][40];
  __shared__ unsigned short Bs[2][128][40];
  v8f acc[2][4] = {};
  const int m0 = blockIdx.x * 128;
  const int n0 = blockIdx.y * 128;
  gemm_mainloop(X, kM, kC, W, m0, n0, As, Bs, acc);

  const int t = threadIdx.x, w = t >> 5, lane = t & 31;
  const int wm = w >> 1, wn = w & 1;
#pragma unroll
  for (int mt = 0; mt < 2; ++mt)
#pragma unroll
    for (int nt = 0; nt < 4; ++nt) {
      int mBase = m0 + wm * 32 + mt * 16 + ((lane >> 4) << 3);
      int j     = n0 + wn * 64 + nt * 16 + (lane & 15);
#pragma unroll
      for (int i = 0; i < 8; ++i) {
        int m = mBase + i;
        if (m >= kM) continue;
        unsigned short bv = f2bf(acc[mt][nt][i]);
        int bb = m / kN, tt = m % kN;
        if (j < 256) {                       // q low-rank
          int h = j >> 4, r = j & 15;
          qcat[(((size_t)bb * kH + h) * kNQ + tt) * kF + r] = bv;
        } else if (j < 512) {                // k low-rank
          int jj = j - 256; int h = jj >> 4, r = jj & 15;
          kcat[(((size_t)bb * kH + h) * kNK + tt) * kF + r] = bv;
        } else if (j < 768) {                // v low-rank (transposed)
          int jj = j - 512; int h = jj >> 4, r = jj & 15;
          vT[(((size_t)bb * kH + h) * kF + r) * kNK + tt] = bv;
        } else if (j < 1792) {               // q full
          int jj = j - 768; int h = jj >> 6, dd = jj & 63;
          qcat[(((size_t)bb * kH + h) * kNQ + tt) * kF + 16 + dd] = bv;
        } else if (j < 2816) {               // k full
          int jj = j - 1792; int h = jj >> 6, dd = jj & 63;
          kcat[(((size_t)bb * kH + h) * kNK + tt) * kF + 16 + dd] = bv;
        } else {                             // v full (transposed)
          int jj = j - 2816; int h = jj >> 6, dd = jj & 63;
          vT[(((size_t)bb * kH + h) * kF + 16 + dd) * kNK + tt] = bv;
        }
      }
    }
}

// ---------------- stage 2: attention core (one block per (b,h)) ------------
__global__ __launch_bounds__(256) void k_attn(
    const unsigned short* __restrict__ qcat,
    const unsigned short* __restrict__ kcat,
    const unsigned short* __restrict__ vT,
    const float* __restrict__ qk_bias,
    unsigned short* __restrict__ xcat) {
  __shared__ float          sc[16][296];     // fp32 scores
  __shared__ unsigned short at[16][312];     // bf16 attn probs (stride 312)
  __shared__ float          qb[kNK];
  __shared__ float          red[16][16];
  __shared__ float          rowinv[16];
  __shared__ float          rowmax[16];

  const int head = blockIdx.x;               // b*16 + h
  const int b    = head >> 4;
  const int h    = head & 15;
  const unsigned short* q = qcat + (size_t)head * kNQ * kF;
  const unsigned short* k = kcat + (size_t)head * kNK * kF;
  const unsigned short* v = vT   + (size_t)head * kF  * kNK;

  for (int j = threadIdx.x; j < kNK; j += 256)
    qb[j] = (j < kN) ? qk_bias[j] : 0.f;
  __syncthreads();

  const int w    = threadIdx.x >> 5;
  const int lane = threadIdx.x & 31;

  for (int qt = 0; qt < 17; ++qt) {
    const int q0 = qt * 16;

    // ---- scores = q_cat . k_cat^T  (waves stride over 18 key tiles) ----
    for (int kt = w; kt < 18; kt += 8) {
      v8f acc = {};
#pragma unroll
      for (int kk = 0; kk < kF; kk += 32) {
        v16bf a  = load_frag_a(q + (size_t)q0 * kF + kk,        kF, lane);
        v16bf bf = load_frag_b(k + (size_t)kt * 16 * kF + kk,   kF, lane);
        acc = wmma_bf16(a, bf, acc);
      }
      int n  = kt * 16 + (lane & 15);
      int mb = (lane >> 4) << 3;
#pragma unroll
      for (int i = 0; i < 8; ++i) sc[mb + i][n] = acc[i];
    }
    __syncthreads();

    // ---- softmax over keys (16 threads per query row) ----
    {
      const int r = threadIdx.x >> 4;
      const int c = threadIdx.x & 15;
      float mx = -1e30f;
      for (int j = c; j < kNK; j += 16) {
        float z = (j < kN) ? (sc[r][j] + qb[j]) * kScale : -1e30f;
        sc[r][j] = z;
        mx = fmaxf(mx, z);
      }
      red[r][c] = mx;
      __syncthreads();
      if (c == 0) {
        float m2 = -1e30f;
#pragma unroll
        for (int i = 0; i < 16; ++i) m2 = fmaxf(m2, red[r][i]);
        rowmax[r] = m2;
      }
      __syncthreads();
      const float rm = rowmax[r];
      float s = 0.f;
      for (int j = c; j < kNK; j += 16) {
        float p = (j < kN) ? __expf(sc[r][j] - rm) : 0.f;
        at[r][j] = f2bf(p);
        s += p;
      }
      red[r][c] = s;
      __syncthreads();
      if (c == 0) {
        float s2 = 0.f;
#pragma unroll
        for (int i = 0; i < 16; ++i) s2 += red[r][i];
        rowinv[r] = 1.f / s2;
      }
      __syncthreads();
    }

    // ---- out = attn . v  (waves 0..5 each own one 16-feature tile) ----
    if (w < 6) {
      const int cbase = w * 16;
      v8f acc = {};
#pragma unroll
      for (int kk = 0; kk < kNK; kk += 32) {
        v16bf a  = load_frag_a(&at[0][0] + kk,                     312, lane);
        v16bf bf = load_frag_b(v + (size_t)cbase * kNK + kk,       kNK, lane);
        acc = wmma_bf16(a, bf, acc);
      }
      int n = cbase + (lane & 15);             // feature 0..95
      if (n < 80) {
        int mb = (lane >> 4) << 3;
#pragma unroll
        for (int i = 0; i < 8; ++i) {
          int qrow = q0 + mb + i;
          if (qrow < kN) {
            float val = acc[i] * rowinv[mb + i];
            size_t m = (size_t)b * kN + qrow;
            xcat[m * kK2 + h * 80 + n] = f2bf(val);
          }
        }
      }
    }
    __syncthreads();
  }
}

// ---------------- stage 3: fused output GEMM (+bias) -----------------------
__global__ __launch_bounds__(256) void k_gemm_out(
    const unsigned short* __restrict__ X, const unsigned short* __restrict__ W,
    const float* __restrict__ bo1, const float* __restrict__ bo2,
    float* __restrict__ out) {
  __shared__ unsigned short As[2][128][40];
  __shared__ unsigned short Bs[2][128][40];
  v8f acc[2][4] = {};
  const int m0 = blockIdx.x * 128;
  const int n0 = blockIdx.y * 128;
  gemm_mainloop(X, kM, kK2, W, m0, n0, As, Bs, acc);

  const int t = threadIdx.x, w = t >> 5, lane = t & 31;
  const int wm = w >> 1, wn = w & 1;
#pragma unroll
  for (int mt = 0; mt < 2; ++mt)
#pragma unroll
    for (int nt = 0; nt < 4; ++nt) {
      int mBase = m0 + wm * 32 + mt * 16 + ((lane >> 4) << 3);
      int o     = n0 + wn * 64 + nt * 16 + (lane & 15);
      float bias = bo1[o] + bo2[o];
#pragma unroll
      for (int i = 0; i < 8; ++i) {
        int m = mBase + i;
        if (m < kM) out[(size_t)m * kC + o] = acc[mt][nt][i] + bias;
      }
    }
}

// ---------------- host launch ----------------------------------------------
extern "C" void kernel_launch(void* const* d_in, const int* in_sizes, int n_in,
                              void* d_out, int out_size, void* d_ws,
                              size_t ws_size, hipStream_t stream) {
  const float* hs      = (const float*)d_in[0];
  const float* Wq_lr   = (const float*)d_in[1];
  const float* Wk_lr   = (const float*)d_in[2];
  const float* Wv_lr   = (const float*)d_in[3];
  const float* Wo_lr   = (const float*)d_in[4];
  const float* bo_lr   = (const float*)d_in[5];
  const float* Wq_full = (const float*)d_in[6];
  const float* Wk_full = (const float*)d_in[7];
  const float* Wv_full = (const float*)d_in[8];
  const float* Wo_full = (const float*)d_in[9];
  const float* bo_full = (const float*)d_in[10];
  const float* qk_bias = (const float*)d_in[11];
  float* out = (float*)d_out;

  char* ws = (char*)d_ws;
  size_t off = 0;
  auto take = [&](size_t bytes) {
    char* p = ws + off;
    off += (bytes + 255) & ~(size_t)255;
    return p;
  };
  unsigned short* hsbf = (unsigned short*)take((size_t)kM * kC * 2);
  unsigned short* wqkv = (unsigned short*)take((size_t)kQKVN * kC * 2);
  unsigned short* wo   = (unsigned short*)take((size_t)kC * kK2 * 2);
  unsigned short* qcat = (unsigned short*)take((size_t)kB * kH * kNQ * kF * 2);
  unsigned short* kcat = (unsigned short*)take((size_t)kB * kH * kNK * kF * 2);
  unsigned short* vT   = (unsigned short*)take((size_t)kB * kH * kF * kNK * 2);
  unsigned short* xcat = (unsigned short*)take((size_t)kM * kK2 * 2);

  // zero q/k/v concat buffers (padding rows/cols must be 0; deterministic)
  {
    long bytes = ((long)kB * kH * kNQ * kF + (long)kB * kH * kNK * kF * 2) * 2;
    long n4 = bytes / 16;
    k_zero_u4<<<(int)((n4 + 255) / 256), 256, 0, stream>>>((uint4*)qcat, n4);
  }
  // convert hidden states to bf16
  {
    long n = (long)kM * kC;
    k_cvt_bf16<<<(int)((n + 255) / 256), 256, 0, stream>>>(hs, hsbf, n);
  }
  // pack fused weights
  {
    long n = (long)kQKVN * kC;
    k_pack_wqkv<<<(int)((n + 255) / 256), 256, 0, stream>>>(
        Wq_lr, Wk_lr, Wv_lr, Wq_full, Wk_full, Wv_full, wqkv);
  }
  {
    long n = (long)kC * kK2;
    k_pack_wo<<<(int)((n + 255) / 256), 256, 0, stream>>>(Wo_lr, Wo_full, wo);
  }
  // stage 1: fused QKV projection GEMM
  k_gemm_qkv<<<dim3((kM + 127) / 128, kQKVN / 128), 256, 0, stream>>>(
      hsbf, wqkv, qcat, kcat, vT);
  // stage 2: attention (one block per (b,h))
  k_attn<<<kB * kH, 256, 0, stream>>>(qcat, kcat, vT, qk_bias, xcat);
  // stage 3: fused output projection GEMM + bias
  k_gemm_out<<<dim3((kM + 127) / 128, kC / 128), 256, 0, stream>>>(
      xcat, wo, bo_lr, bo_full, out);
}
